// MinkowskiConvolution_19155554140408
// MI455X (gfx1250) — matrix operand order repaired and checked by reading, not compiled, not run on hardware
//
#include <hip/hip_runtime.h>

typedef __attribute__((ext_vector_type(16))) _Float16 v16h;
typedef __attribute__((ext_vector_type(8)))  _Float16 v8h;
typedef __attribute__((ext_vector_type(8)))  float    v8f;
typedef __attribute__((ext_vector_type(4)))  float    v4f;
typedef __attribute__((ext_vector_type(4)))  unsigned int v4u;

#define KK   27
#define INC  32
#define OUTC 64
#define NGRP 4                       // OUTC / 16
#define B_TILE_HALFS (32 * 16)       // one 32x16 f16 B tile: 32 lanes x 16 f16
#define B_TOTAL_HALFS (KK * NGRP * B_TILE_HALFS)   // 55296 f16 = 110592 B
#define MAX_BLOCKS 768               // persistent blocks (grid-stride)

// ---------------------------------------------------------------------------
// One-time cast: features fp32 -> f16, row-major [N][32] (64 B per row).
// ---------------------------------------------------------------------------
__global__ void feat_f32_to_f16(const float* __restrict__ src,
                                _Float16* __restrict__ dst, int n) {
    int i = blockIdx.x * blockDim.x + threadIdx.x;
    int base = i * 8;
    if (base + 8 <= n) {
        v4f f0 = ((const v4f*)(src + base))[0];
        v4f f1 = ((const v4f*)(src + base))[1];
        v8h h;
        h[0] = (_Float16)f0[0]; h[1] = (_Float16)f0[1];
        h[2] = (_Float16)f0[2]; h[3] = (_Float16)f0[3];
        h[4] = (_Float16)f1[0]; h[5] = (_Float16)f1[1];
        h[6] = (_Float16)f1[2]; h[7] = (_Float16)f1[3];
        *(v8h*)(dst + base) = h;
    } else {
        for (int j = base; j < n; ++j) dst[j] = (_Float16)src[j];
    }
}

// ---------------------------------------------------------------------------
// One-time swizzle of weights [27][32][64] fp32 into WMMA B-operand layout,
// f16; also zeroes the 32-half "zero row" used for invalid-neighbor gathers.
// Tile t = k*4+g (32x16: K=inc, N=16 out channels):
//   lane l: column o = l&15, K-half = l>>4; v16h element e = B[(e>>3)*16+half*8+(e&7)][o].
// ---------------------------------------------------------------------------
__global__ void prep_B(const float* __restrict__ w, _Float16* __restrict__ Bpre,
                       _Float16* __restrict__ zrow) {
    int t = blockIdx.x * blockDim.x + threadIdx.x;
    if (t < KK * NGRP * 32) {
        int lane  = t & 31;
        int tg    = t >> 5;           // k*4 + g
        int k     = tg >> 2;
        int g     = tg & 3;
        int col   = lane & 15;
        int khalf = lane >> 4;
        const float* wk = w + (size_t)k * INC * OUTC + (size_t)(g * 16 + col);
        v16h bv;
        for (int e = 0; e < 16; ++e)
            bv[e] = (_Float16)wk[(size_t)((e >> 3) * 16 + khalf * 8 + (e & 7)) * OUTC];
        ((v16h*)Bpre)[t] = bv;
    } else if (t < KK * NGRP * 32 + 32) {
        zrow[t - KK * NGRP * 32] = (_Float16)0.0f;   // 64-B zero row
    }
}

// ---------------------------------------------------------------------------
// A-operand gather for one 16-row tile: lane<16 -> K 0-7,16-23 at byte offs
// {0,32}; lane>=16 -> K 8-15,24-31 at byte offs {16,48}. Invalid neighbors
// redirect the pointer to a shared zero row (single 64-bit cndmask; all
// invalid lanes hit one cacheline). EXEC stays all-1s for WMMA.
// ---------------------------------------------------------------------------
__device__ __forceinline__ v16h gather_A(const _Float16* __restrict__ feat16,
                                         const _Float16* __restrict__ zrow,
                                         int idx, int kh) {
    const _Float16* p = (idx >= 0) ? (feat16 + (size_t)idx * INC) : zrow;
    const v4u* rowp = (const v4u*)p;
    union { v4u u[2]; v16h h; } ua;
    ua.u[0] = rowp[kh];
    ua.u[1] = rowp[2 + kh];
    return ua.h;
}

// ---------------------------------------------------------------------------
// Main kernel: persistent blocks, 8 wave32s each. Block stages all B tiles
// in LDS once; each wave grid-strides over 32-row output pairs. Per k-offset:
// 8 ds_load_b128 -> 8 v_wmma_f32_16x16x32_f16 (two A tiles share each B).
// ---------------------------------------------------------------------------
__global__ __launch_bounds__(256) void spconv_wmma(
    const _Float16* __restrict__ feat16,
    const _Float16* __restrict__ Bpre,
    const _Float16* __restrict__ zrow,
    const int* __restrict__ nidx,
    float* __restrict__ out,
    int N, int npairs)
{
    __shared__ __align__(32) _Float16 Bs[B_TOTAL_HALFS];

    {   // cooperative LDS fill: 6912 x 16B chunks (once per persistent block)
        const v4u* s = (const v4u*)Bpre;
        v4u* d = (v4u*)Bs;
        const int chunks = B_TOTAL_HALFS / 8;
        for (int i = threadIdx.x; i < chunks; i += blockDim.x) d[i] = s[i];
    }
    __syncthreads();

    const int wave   = threadIdx.x >> 5;
    const int lane   = threadIdx.x & 31;
    const int col    = lane & 15;        // A row / B,D column within tile
    const int kh     = lane >> 4;        // K-half select
    const int waveId = blockIdx.x * (blockDim.x >> 5) + wave;
    const int nWaves = gridDim.x * (blockDim.x >> 5);

    for (int pair = waveId; pair < npairs; pair += nWaves) {
        const int n0 = pair * 32;        // rows [n0, n0+32)
        int r0 = n0 + col;       if (r0 >= N) r0 = N - 1;
        int r1 = n0 + 16 + col;  if (r1 >= N) r1 = N - 1;

        v8f acc0[NGRP] = {{}, {}, {}, {}};
        v8f acc1[NGRP] = {{}, {}, {}, {}};

        const int* ip0 = nidx + r0;
        const int* ip1 = nidx + r1;

        int idx0 = ip0[0];
        int idx1 = ip1[0];
        for (int k = 0; k < KK; ++k) {
            v16h A0 = gather_A(feat16, zrow, idx0, kh);
            v16h A1 = gather_A(feat16, zrow, idx1, kh);
            if (k + 1 < KK) {            // prefetch next offsets' indices
                idx0 = ip0[(size_t)(k + 1) * N];
                idx1 = ip1[(size_t)(k + 1) * N];
            }
            const v16h* bs = (const v16h*)(Bs + (size_t)k * NGRP * B_TILE_HALFS);
#pragma unroll
            for (int g = 0; g < NGRP; ++g) {
                v16h B = bs[g * 32 + lane];  // 2 x ds_load_b128
                acc0[g] = __builtin_amdgcn_wmma_f32_16x16x32_f16(
                              false, A0, false, B, (short)0, acc0[g], false, false);
                acc1[g] = __builtin_amdgcn_wmma_f32_16x16x32_f16(
                              false, A1, false, B, (short)0, acc1[g], false, false);
            }
        }

        // D layout: VGPR v -> row M = v + 8*kh, lane -> column (l&15).
#pragma unroll
        for (int t = 0; t < 2; ++t) {
            float* op = out + (size_t)(n0 + t * 16) * OUTC;
#pragma unroll
            for (int v = 0; v < 8; ++v) {
                int m = v + 8 * kh;
                if (n0 + t * 16 + m < N) {
#pragma unroll
                    for (int g = 0; g < NGRP; ++g)
                        op[(size_t)m * OUTC + g * 16 + col] =
                            (t == 0) ? acc0[g][v] : acc1[g][v];
                }
            }
        }
    }
}

// ---------------------------------------------------------------------------
extern "C" void kernel_launch(void* const* d_in, const int* in_sizes, int n_in,
                              void* d_out, int out_size, void* d_ws, size_t ws_size,
                              hipStream_t stream) {
    const float* features = (const float*)d_in[0];   // [N,32] fp32
    const float* kernelw  = (const float*)d_in[1];   // [27,32,64] fp32
    const int*   nidx     = (const int*)d_in[2];     // [27,N] int32
    float* out = (float*)d_out;                      // [N,64] fp32

    const int N = in_sizes[0] / INC;

    // Workspace: f16 feature table, pre-swizzled f16 B tiles, 64-B zero row.
    // Needs N*32*2 (19.2 MB for N=300k) + ~110.7 KB.
    size_t featBytes = ((size_t)N * INC * sizeof(_Float16) + 255) & ~(size_t)255;
    size_t bBytes    = ((size_t)B_TOTAL_HALFS * sizeof(_Float16) + 255) & ~(size_t)255;
    _Float16* feat16 = (_Float16*)d_ws;
    _Float16* Bpre   = (_Float16*)((char*)d_ws + featBytes);
    _Float16* zrow   = (_Float16*)((char*)d_ws + featBytes + bBytes);

    const int nElems = N * INC;
    const int t8 = (nElems + 7) / 8;
    feat_f32_to_f16<<<(t8 + 255) / 256, 256, 0, stream>>>(features, feat16, nElems);

    prep_B<<<(KK * NGRP * 32 + 32 + 255) / 256, 256, 0, stream>>>(kernelw, Bpre, zrow);

    const int npairs = (N + 31) / 32;                // 32 rows per wave-pass
    const int wavesPerBlock = 8;                     // 256 threads, wave32
    int blocks = (npairs + wavesPerBlock - 1) / wavesPerBlock;
    if (blocks > MAX_BLOCKS) blocks = MAX_BLOCKS;    // persistent grid-stride
    spconv_wmma<<<blocks, 256, 0, stream>>>(feat16, Bpre, zrow, nidx, out, N, npairs);
}
